// DemoBlock_55018531061930
// MI455X (gfx1250) — 8-bit matrix operand order repaired and checked by reading, compiled, not run on hardware
//
#include <hip/hip_runtime.h>
#include <hip/hip_bf16.h>
#include <math.h>
#include <stdint.h>


typedef _Float16 half_t;
typedef __attribute__((ext_vector_type(8)))  _Float16 v8h;
typedef __attribute__((ext_vector_type(8)))  float    v8f;
typedef __attribute__((ext_vector_type(2)))  int      v2i;
typedef __attribute__((ext_vector_type(4)))  int      v4i;
typedef __attribute__((ext_vector_type(16))) int      v16i;

#define D_MODEL 1024
#define S_LEN   4096
#define BATCH   4
#define ROWS    (BATCH * S_LEN)
#define N_QKV   (3 * D_MODEL)

#define C3_SCALE 0.04953007815f
#define ROPE_L2  0.025952562419f
#define W_SCALE   64.0f
#define W_ISCALE  (1.0f / 64.0f)

__device__ __forceinline__ v8f vzero8() {
  v8f r;
#pragma unroll
  for (int i = 0; i < 8; ++i) r[i] = 0.f;
  return r;
}

__device__ __forceinline__ void async_g2l_b128(const void* gptr, void* lptr) {
  unsigned lofs = (unsigned)(uintptr_t)lptr;
  unsigned long long ga = (unsigned long long)(uintptr_t)gptr;
  asm volatile("global_load_async_to_lds_b128 %0, %1, off"
               :: "v"(lofs), "v"(ga) : "memory");
}
__device__ __forceinline__ void wait_async0() {
  asm volatile("s_wait_asynccnt 0x0" ::: "memory");
}

__device__ __forceinline__ unsigned char f32_to_e4m3_sw(float x) {
  unsigned int u = __float_as_uint(x);
  unsigned char sign = (unsigned char)((u >> 24) & 0x80);
  float a = fabsf(x);
  if (a > 448.f) a = 448.f;
  if (a < 0.0009765625f) return sign;
  int e;
  float m = frexpf(a, &e);
  int E = e - 1 + 7;
  int mant;
  if (E <= 0) {
    mant = (int)rintf(a * 512.0f);
    if (mant >= 8) { E = 1; mant = 0; }
    else return sign | (unsigned char)mant;
  } else {
    mant = (int)rintf((m * 2.0f - 1.0f) * 8.0f);
    if (mant == 8) { mant = 0; E += 1; }
    if (E >= 16) return sign | 0x7E;
  }
  return sign | (unsigned char)((E << 3) | mant);
}
__device__ __forceinline__ unsigned char cvt_e4m3(float f) {
#if __has_builtin(__builtin_amdgcn_cvt_pk_fp8_f32)
  return (unsigned char)(__builtin_amdgcn_cvt_pk_fp8_f32(f, f, 0, false) & 0xff);
#else
  return f32_to_e4m3_sw(f);
#endif
}
__device__ __forceinline__ unsigned int pack4_e4m3(float f0, float f1, float f2, float f3) {
#if __has_builtin(__builtin_amdgcn_cvt_pk_fp8_f32)
  int v = __builtin_amdgcn_cvt_pk_fp8_f32(f0, f1, 0, false);
  v = __builtin_amdgcn_cvt_pk_fp8_f32(f2, f3, v, true);
  return (unsigned int)v;
#else
  return (unsigned)f32_to_e4m3_sw(f0) | ((unsigned)f32_to_e4m3_sw(f1) << 8) |
         ((unsigned)f32_to_e4m3_sw(f2) << 16) | ((unsigned)f32_to_e4m3_sw(f3) << 24);
#endif
}

__device__ __forceinline__ v16i frag_a8(const unsigned char* base, int strideB,
                                        int kb, int lane) {
  const unsigned char* p = base + (size_t)(lane & 15) * strideB + kb + (lane >> 4) * 16;
  v16i r;
#pragma unroll
  for (int blk = 0; blk < 2; ++blk)
#pragma unroll
    for (int j = 0; j < 4; ++j) {
      v2i t = *(const v2i*)(p + blk * 64 + (j >> 1) * 32 + (j & 1) * 8);
      r[blk * 8 + 2 * j]     = t[0];
      r[blk * 8 + 2 * j + 1] = t[1];
    }
  return r;
}
__device__ __forceinline__ v16i frag_b8(const unsigned char* base, size_t strideB,
                                        int nOff, int kb, int lane) {
  const unsigned char* p = base + (size_t)(nOff + (lane & 15)) * strideB + kb
                         + (lane >> 4) * 16;
  v16i r;
#pragma unroll
  for (int s = 0; s < 4; ++s) {
    v4i t = *(const v4i*)(p + s * 32);
    r[4 * s] = t[0]; r[4 * s + 1] = t[1]; r[4 * s + 2] = t[2]; r[4 * s + 3] = t[3];
  }
  return r;
}
__device__ __forceinline__ v8f wmma8(v16i a, v16i b, v8f c) {
  return __builtin_amdgcn_wmma_f32_16x16x128_fp8_fp8(a, b, (short)0, c, false, false);
}

__global__ void pack_wt8_kernel(const float* __restrict__ w, unsigned char* __restrict__ wt8,
                                int K, int N) {
  const size_t id = (size_t)blockIdx.x * blockDim.x + threadIdx.x;
  if (id >= (size_t)K * N) return;
  const int n = (int)(id / K);
  const int k = (int)(id % K);
  wt8[id] = cvt_e4m3(w[(size_t)k * N + n] * W_SCALE);
}

__global__ __launch_bounds__(256) void rmsnorm8_kernel(const float* __restrict__ x,
                                                       const float* __restrict__ g,
                                                       unsigned char* __restrict__ h8) {
  __shared__ float red[256];
  const int row = blockIdx.x;
  const int tid = threadIdx.x;
  const float* xr = x + (size_t)row * D_MODEL;
  float v[4];
  float s = 0.f;
#pragma unroll
  for (int i = 0; i < 4; ++i) { v[i] = xr[tid + i * 256]; s += v[i] * v[i]; }
  red[tid] = s;
  __syncthreads();
  for (int off = 128; off > 0; off >>= 1) {
    if (tid < off) red[tid] += red[tid + off];
    __syncthreads();
  }
  const float rs = rsqrtf(red[0] * (1.0f / D_MODEL) + 1e-6f);
#pragma unroll
  for (int i = 0; i < 4; ++i)
    h8[(size_t)row * D_MODEL + tid + i * 256] = cvt_e4m3(v[i] * rs * g[tid + i * 256]);
}

__global__ __launch_bounds__(128) void gemm8_qkv_kernel(const unsigned char* __restrict__ A8,
                                                        const unsigned char* __restrict__ Bt8,
                                                        const float* __restrict__ bias,
                                                        half_t* __restrict__ C) {
  __shared__ __align__(16) unsigned char As[2][64][144];
  const int tid = threadIdx.x, lane = tid & 31, wave = tid >> 5;
  const int wr = wave >> 1, wc = wave & 1, hi = lane >> 4;
  const int m0 = blockIdx.x * 64, n0 = blockIdx.y * 64;
  const int r = tid >> 1, cb = (tid & 1) * 64;
  const unsigned char* arow  = A8 + (size_t)(m0 + r) * D_MODEL + cb;
  const unsigned char* btile = Bt8 + (size_t)(n0 + wc * 32) * D_MODEL;

  v8f acc[2][2];
#pragma unroll
  for (int i = 0; i < 2; ++i)
#pragma unroll
    for (int j = 0; j < 2; ++j) acc[i][j] = vzero8();

#pragma unroll
  for (int i = 0; i < 4; ++i)
    async_g2l_b128(arow + i * 16, &As[0][r][cb + i * 16]);

#pragma unroll
  for (int it = 0; it < 8; ++it) {
    const int buf = it & 1;
    wait_async0();
    __syncthreads();
    if (it < 7) {
#pragma unroll
      for (int i = 0; i < 4; ++i)
        async_g2l_b128(arow + (it + 1) * 128 + i * 16, &As[buf ^ 1][r][cb + i * 16]);
    }
    const int kb = it * 128;
    v16i a0 = frag_a8(&As[buf][wr * 32][0],      144, 0, lane);
    v16i a1 = frag_a8(&As[buf][wr * 32 + 16][0], 144, 0, lane);
    v16i b0 = frag_b8(btile, D_MODEL, 0,  kb, lane);
    v16i b1 = frag_b8(btile, D_MODEL, 16, kb, lane);
    acc[0][0] = wmma8(a0, b0, acc[0][0]);
    acc[0][1] = wmma8(a0, b1, acc[0][1]);
    acc[1][0] = wmma8(a1, b0, acc[1][0]);
    acc[1][1] = wmma8(a1, b1, acc[1][1]);
  }

#pragma unroll
  for (int i2 = 0; i2 < 2; ++i2)
#pragma unroll
    for (int j2 = 0; j2 < 2; ++j2) {
      const int n = n0 + wc * 32 + j2 * 16 + (lane & 15);
      const float bb = bias[n];
#pragma unroll
      for (int i = 0; i < 8; ++i) {
        const int row = m0 + wr * 32 + i2 * 16 + hi * 8 + i;
        C[(size_t)row * N_QKV + n] = (half_t)(acc[i2][j2][i] * W_ISCALE + bb);
      }
    }
}

__global__ void rope_quant_kernel(const half_t* __restrict__ qkv,
                                  unsigned char* __restrict__ qf8,
                                  unsigned char* __restrict__ kf8) {
  const size_t id = (size_t)blockIdx.x * blockDim.x + threadIdx.x;
  const int j   = (int)(id & 511);
  const int qk  = (int)((id >> 9) & 1);
  const int row = (int)(id >> 10);
  const int s   = row & (S_LEN - 1);
  const half_t* p = qkv + (size_t)row * N_QKV + qk * D_MODEL;
  unsigned char* dst = (qk ? kf8 : qf8) + (size_t)row * D_MODEL;
  const float inv = exp2f(-(float)j * ROPE_L2);
  const float ang = (float)s * inv;
  const float c  = rintf(cosf(ang));
  const float sn = rintf(sinf(ang));
  const float t1 = (float)p[j];
  const float t2 = (float)p[j + 512];
  dst[j]       = cvt_e4m3(t1 * c - t2 * sn);
  dst[j + 512] = cvt_e4m3(t2 * c + t1 * sn);
}

__global__ __launch_bounds__(256) void transpose_v8_kernel(const half_t* __restrict__ qkv,
                                                           unsigned char* __restrict__ vt8) {
  __shared__ __align__(16) half_t tile[64][72];
  const int b  = blockIdx.z;
  const int s0 = blockIdx.x * 64;
  const int d0 = blockIdx.y * 64;
  const int t  = threadIdx.x;
  const int r = t >> 2, cb = (t & 3) * 16;
  const half_t* src = qkv + ((size_t)(b * S_LEN + s0 + r)) * N_QKV + 2 * D_MODEL + d0 + cb;
  *(v8h*)&tile[r][cb]     = *(const v8h*)src;
  *(v8h*)&tile[r][cb + 8] = *(const v8h*)(src + 8);
  __syncthreads();
  unsigned char* dst = vt8 + ((size_t)b * D_MODEL + d0 + r) * S_LEN + s0 + cb;
  v4i w;
#pragma unroll
  for (int c = 0; c < 4; ++c)
    w[c] = (int)pack4_e4m3((float)tile[cb + 4 * c + 0][r], (float)tile[cb + 4 * c + 1][r],
                           (float)tile[cb + 4 * c + 2][r], (float)tile[cb + 4 * c + 3][r]);
  *(v4i*)dst = w;
}

__global__ __launch_bounds__(128) void attn_kernel(const unsigned char* __restrict__ qf8,
                                                   const unsigned char* __restrict__ kf8,
                                                   const unsigned char* __restrict__ vt8,
                                                   unsigned char* __restrict__ o8) {
  __shared__ __align__(16) unsigned char Qs[16][1040];
  __shared__ __align__(16) float         Sacc[16][128];
  __shared__ __align__(16) unsigned char Ps[16][144];
  __shared__ float pmax[16][8], psum[16][8];
  __shared__ float mrow[16], lrow[16], arow[16];

  const int tid  = threadIdx.x;
  const int lane = tid & 31;
  const int wave = tid >> 5;
  const int hi   = lane >> 4;
  const int b    = blockIdx.x >> 8;
  const int q0   = (blockIdx.x & 255) * 16;

  {
    const int r  = tid >> 3;
    const int cbb = (tid & 7) * 128;
    const unsigned char* src = qf8 + ((size_t)(b * S_LEN + q0 + r)) * D_MODEL + cbb;
#pragma unroll
    for (int i = 0; i < 8; ++i)
      async_g2l_b128(src + i * 16, &Qs[r][cbb + i * 16]);
    wait_async0();
  }
  if (tid < 16) { mrow[tid] = -1e30f; lrow[tid] = 0.f; }

  v8f oacc[16];
#pragma unroll
  for (int t = 0; t < 16; ++t) oacc[t] = vzero8();

  const unsigned char* vtb = vt8 + (size_t)b * D_MODEL * S_LEN;
  __syncthreads();

  for (int kt = 0; kt < S_LEN / 128; ++kt) {
    const unsigned char* kbase = kf8 + ((size_t)(b * S_LEN + kt * 128)) * D_MODEL;
    v8f sc0 = vzero8(), sc1 = vzero8();
#pragma unroll
    for (int st = 0; st < 8; ++st) {
      const int kb = st * 128;
      v16i a  = frag_a8(&Qs[0][0], 1040, kb, lane);
      v16i b0 = frag_b8(kbase, D_MODEL, wave * 32,      kb, lane);
      v16i b1 = frag_b8(kbase, D_MODEL, wave * 32 + 16, kb, lane);
      sc0 = wmma8(a, b0, sc0);
      sc1 = wmma8(a, b1, sc1);
    }
    if (kt + 1 < S_LEN / 128) {
      __builtin_prefetch(kbase + (size_t)128 * D_MODEL + (size_t)lane * D_MODEL, 0, 1);
      __builtin_prefetch(vtb + (size_t)(wave * 256 + lane) * S_LEN + (kt + 1) * 128, 0, 1);
    }
    {
      const int c0 = wave * 32 + (lane & 15);
#pragma unroll
      for (int i = 0; i < 8; ++i) {
        Sacc[hi * 8 + i][c0]      = sc0[i];
        Sacc[hi * 8 + i][c0 + 16] = sc1[i];
      }
    }
    __syncthreads();

    {
      const int r = tid >> 3, c0 = (tid & 7) * 16;
      float mx = -1e30f;
#pragma unroll
      for (int j = 0; j < 16; ++j) mx = fmaxf(mx, Sacc[r][c0 + j]);
      pmax[r][tid & 7] = mx;
    }
    __syncthreads();
    if (tid < 16) {
      float mx = mrow[tid];
#pragma unroll
      for (int j = 0; j < 8; ++j) mx = fmaxf(mx, pmax[tid][j]);
      arow[tid] = exp2f((mrow[tid] - mx) * C3_SCALE);
      mrow[tid] = mx;
    }
    __syncthreads();
    {
      const int r = tid >> 3, c0 = (tid & 7) * 16;
      const float mx = mrow[r];
      float s = 0.f;
#pragma unroll
      for (int j = 0; j < 16; ++j) {
        const float p = exp2f((Sacc[r][c0 + j] - mx) * C3_SCALE);
        Ps[r][c0 + j] = cvt_e4m3(p);
        s += p;
      }
      psum[r][tid & 7] = s;
    }
    __syncthreads();
    if (tid < 16) {
      float s = 0.f;
#pragma unroll
      for (int j = 0; j < 8; ++j) s += psum[tid][j];
      lrow[tid] = lrow[tid] * arow[tid] + s;
    }
    __syncthreads();

    float al[8];
#pragma unroll
    for (int i = 0; i < 8; ++i) al[i] = arow[hi * 8 + i];
    v16i pa = frag_a8(&Ps[0][0], 144, 0, lane);
#pragma unroll
    for (int t = 0; t < 16; ++t) {
      v16i vb = frag_b8(vtb, S_LEN, wave * 256 + t * 16, kt * 128, lane);
#pragma unroll
      for (int i = 0; i < 8; ++i) oacc[t][i] *= al[i];
      oacc[t] = wmma8(pa, vb, oacc[t]);
    }
    __syncthreads();
  }

  float linv[8];
#pragma unroll
  for (int i = 0; i < 8; ++i) linv[i] = 1.f / lrow[hi * 8 + i];
#pragma unroll
  for (int t = 0; t < 16; ++t) {
    const int n = wave * 256 + t * 16 + (lane & 15);
#pragma unroll
    for (int i = 0; i < 8; ++i) {
      const int r = hi * 8 + i;
      o8[((size_t)(b * S_LEN + q0 + r)) * D_MODEL + n] = cvt_e4m3(oacc[t][i] * linv[i]);
    }
  }
}

__global__ __launch_bounds__(128) void gemm8_proj_kernel(const unsigned char* __restrict__ A8,
                                                         const unsigned char* __restrict__ Bt8,
                                                         const float* __restrict__ bias,
                                                         const float* __restrict__ resid,
                                                         float* __restrict__ out) {
  __shared__ __align__(16) unsigned char As[2][64][144];
  const int tid = threadIdx.x, lane = tid & 31, wave = tid >> 5;
  const int wr = wave >> 1, wc = wave & 1, hi = lane >> 4;
  const int m0 = blockIdx.x * 64, n0 = blockIdx.y * 64;
  const int r = tid >> 1, cb = (tid & 1) * 64;
  const unsigned char* arow  = A8 + (size_t)(m0 + r) * D_MODEL + cb;
  const unsigned char* btile = Bt8 + (size_t)(n0 + wc * 32) * D_MODEL;

  v8f acc[2][2];
#pragma unroll
  for (int i = 0; i < 2; ++i)
#pragma unroll
    for (int j = 0; j < 2; ++j) acc[i][j] = vzero8();

#pragma unroll
  for (int i = 0; i < 4; ++i)
    async_g2l_b128(arow + i * 16, &As[0][r][cb + i * 16]);

#pragma unroll
  for (int it = 0; it < 8; ++it) {
    const int buf = it & 1;
    wait_async0();
    __syncthreads();
    if (it < 7) {
#pragma unroll
      for (int i = 0; i < 4; ++i)
        async_g2l_b128(arow + (it + 1) * 128 + i * 16, &As[buf ^ 1][r][cb + i * 16]);
    }
    const int kb = it * 128;
    v16i a0 = frag_a8(&As[buf][wr * 32][0],      144, 0, lane);
    v16i a1 = frag_a8(&As[buf][wr * 32 + 16][0], 144, 0, lane);
    v16i b0 = frag_b8(btile, D_MODEL, 0,  kb, lane);
    v16i b1 = frag_b8(btile, D_MODEL, 16, kb, lane);
    acc[0][0] = wmma8(a0, b0, acc[0][0]);
    acc[0][1] = wmma8(a0, b1, acc[0][1]);
    acc[1][0] = wmma8(a1, b0, acc[1][0]);
    acc[1][1] = wmma8(a1, b1, acc[1][1]);
  }

#pragma unroll
  for (int i2 = 0; i2 < 2; ++i2)
#pragma unroll
    for (int j2 = 0; j2 < 2; ++j2) {
      const int n = n0 + wc * 32 + j2 * 16 + (lane & 15);
      const float bb = bias[n];
#pragma unroll
      for (int i = 0; i < 8; ++i) {
        const int row = m0 + wr * 32 + i2 * 16 + hi * 8 + i;
        out[(size_t)row * D_MODEL + n] =
            acc[i2][j2][i] * W_ISCALE + bb + resid[(size_t)row * D_MODEL + n];
      }
    }
}

extern "C" void kernel_launch(void* const* d_in, const int* in_sizes, int n_in,
                              void* d_out, int out_size, void* d_ws, size_t ws_size,
                              hipStream_t stream) {
  (void)in_sizes; (void)n_in; (void)out_size; (void)ws_size;
  const float* x     = (const float*)d_in[0];
  const float* g     = (const float*)d_in[1];
  const float* wqkv  = (const float*)d_in[2];
  const float* bqkv  = (const float*)d_in[3];
  const float* wproj = (const float*)d_in[4];
  const float* bproj = (const float*)d_in[5];
  float* out = (float*)d_out;

  char* ws = (char*)d_ws;
  half_t*        QKV  = (half_t*)(ws + ((size_t)0   << 20));
  unsigned char* H8   = (unsigned char*)(ws + ((size_t)96  << 20));
  unsigned char* QF8  = (unsigned char*)(ws + ((size_t)112 << 20));
  unsigned char* KF8  = (unsigned char*)(ws + ((size_t)128 << 20));
  unsigned char* VT8  = (unsigned char*)(ws + ((size_t)144 << 20));
  unsigned char* O8   = (unsigned char*)(ws + ((size_t)160 << 20));
  unsigned char* WQT8 = (unsigned char*)(ws + ((size_t)176 << 20));
  unsigned char* WPT8 = (unsigned char*)(ws + ((size_t)179 << 20));

  pack_wt8_kernel<<<(N_QKV * D_MODEL + 255) / 256, 256, 0, stream>>>(wqkv, WQT8, D_MODEL, N_QKV);
  pack_wt8_kernel<<<(D_MODEL * D_MODEL + 255) / 256, 256, 0, stream>>>(wproj, WPT8, D_MODEL, D_MODEL);
  rmsnorm8_kernel<<<ROWS, 256, 0, stream>>>(x, g, H8);
  gemm8_qkv_kernel<<<dim3(ROWS / 64, N_QKV / 64), 128, 0, stream>>>(H8, WQT8, bqkv, QKV);
  rope_quant_kernel<<<(ROWS * 2 * 512) / 256, 256, 0, stream>>>(QKV, QF8, KF8);
  transpose_v8_kernel<<<dim3(S_LEN / 64, D_MODEL / 64, BATCH), 256, 0, stream>>>(QKV, VT8);
  attn_kernel<<<BATCH * (S_LEN / 16), 128, 0, stream>>>(QF8, KF8, VT8, O8);
  gemm8_proj_kernel<<<dim3(ROWS / 64, D_MODEL / 64), 128, 0, stream>>>(O8, WPT8, bqkv ? bproj : bproj, x, out);
}
